// Nemotron3Block_43542378447403
// MI455X (gfx1250) — matrix operand order repaired and checked by reading, not compiled, not run on hardware
//
#include <hip/hip_runtime.h>
#include <math.h>

// ---------------- dims ----------------
#define S_LEN   2048
#define D_MODEL 4096
#define H_M     128
#define P_M     64
#define G_M     8
#define N_M     128
#define D_INNER 8192              // H_M*P_M
#define GN      1024              // G_M*N_M
#define CONV_DIM 10240            // D_INNER + 2*GN
#define PROJ_DIM 18560            // D_INNER + CONV_DIM + H_M
#define CHUNK   128
#define NCHUNK  16                // S_LEN/CHUNK
#define EPSF    1e-5f

typedef __attribute__((ext_vector_type(16))) __bf16 v16bf;
typedef __attribute__((ext_vector_type(8)))  __bf16 v8bf;
typedef __attribute__((ext_vector_type(8)))  float  v8f;
typedef __attribute__((ext_vector_type(4)))  unsigned v4u;
typedef __attribute__((ext_vector_type(8)))  int      v8i;
typedef __attribute__((ext_vector_type(4)))  int      v4i;

__device__ __forceinline__ v16bf cat16(v8bf lo, v8bf hi) {
  return __builtin_shufflevector(lo, hi, 0,1,2,3,4,5,6,7,8,9,10,11,12,13,14,15);
}
// A-matrix fragment: lane (m = lane&15, lh = lane>>4) holds K = {8lh..8lh+7, 16+8lh..16+8lh+7}
__device__ __forceinline__ v16bf load_afrag(const __bf16* row_k, int lh) {
  v8bf lo = *(const v8bf*)(row_k + 8*lh);
  v8bf hi = *(const v8bf*)(row_k + 16 + 8*lh);
  return cat16(lo, hi);
}
// B-matrix fragment: lane (n = lane&15, lh = lane>>4) holds K = {8lh..8lh+7, 16+8lh..16+8lh+7}
__device__ __forceinline__ v16bf load_bfrag(const __bf16* col_k, int lh) {
  v8bf lo = *(const v8bf*)(col_k + 8*lh);
  v8bf hi = *(const v8bf*)(col_k + 16 + 8*lh);
  return cat16(lo, hi);
}
__device__ __forceinline__ v8f wmma_bf16(v16bf a, v16bf b, v8f c) {
  return __builtin_amdgcn_wmma_f32_16x16x32_bf16(false, a, false, b, (short)0, c, false, false);
}
__device__ __forceinline__ float sigmoidf_(float x) { return 1.0f / (1.0f + __expf(-x)); }

// ---------------- Tensor Data Mover: 2-D bf16 tile DMA global -> LDS ----------------
// D# per CDNA5 ISA ch.8: group0 = {count, lds_addr, global_addr(57b), type=2},
// group1 = {wg_mask=0, data_size, tensor_dim0/1, tile_dim0/1/2, dim0_stride}.
__device__ __forceinline__ void tdm_load_2d_bf16(unsigned lds_addr, const void* gptr,
                                                 unsigned tile0, unsigned tile1,
                                                 unsigned td0, unsigned td1,
                                                 unsigned stride0) {
  const unsigned long long ga = (unsigned long long)gptr;
  v4u g0;
  g0[0] = 1u;                                                   // count=1, user mode
  g0[1] = lds_addr;                                             // LDS byte address
  g0[2] = (unsigned)(ga & 0xFFFFFFFFu);                         // global_addr[31:0]
  g0[3] = (unsigned)((ga >> 32) & 0x01FFFFFFu) | (2u << 30);    // global_addr[56:32] | type=2
  v8i g1;
  g1[0] = (int)(1u << 16);                                      // wg_mask=0, data_size=1 (2B)
  g1[1] = (int)((td0 & 0xFFFFu) << 16);                         // tensor_dim0[15:0]
  g1[2] = (int)((td0 >> 16) | ((td1 & 0xFFFFu) << 16));         // td0[31:16] | td1[15:0]
  g1[3] = (int)((td1 >> 16) | ((tile0 & 0xFFFFu) << 16));       // td1[31:16] | tile_dim0
  g1[4] = (int)(tile1 & 0xFFFFu);                               // tile_dim1, tile_dim2=0
  g1[5] = (int)stride0;                                         // tensor_dim0_stride[31:0]
  g1[6] = 0;                                                    // stride0[47:32]=0, dim1_stride lo=0
  g1[7] = 0;
#if __has_builtin(__builtin_amdgcn_tensor_load_to_lds)
  v4i z4 = 0;
  v8i z8 = 0;
  __builtin_amdgcn_tensor_load_to_lds(g0, g1, z4, z4, z8, 0);
#else
  asm volatile("tensor_load_to_lds %0, %1" :: "s"(g0), "s"(g1) : "memory");
#endif
}
__device__ __forceinline__ void wait_tensorcnt0() {
#if __has_builtin(__builtin_amdgcn_s_wait_tensorcnt)
  __builtin_amdgcn_s_wait_tensorcnt((short)0);
#else
  asm volatile("s_wait_tensorcnt 0x0" ::: "memory");
#endif
}
__device__ __forceinline__ void wait_tensorcnt1() {
#if __has_builtin(__builtin_amdgcn_s_wait_tensorcnt)
  __builtin_amdgcn_s_wait_tensorcnt((short)1);
#else
  asm volatile("s_wait_tensorcnt 0x1" ::: "memory");
#endif
}

// ---------------- K1: RMSNorm + cast to bf16 ----------------
__global__ void rmsnorm_cast_kernel(const float* __restrict__ x, const float* __restrict__ w,
                                    __bf16* __restrict__ h) {
  __shared__ float red[256];
  const int s = blockIdx.x;
  const float* xr = x + (size_t)s * D_MODEL;
  float ss = 0.f;
  for (int d = threadIdx.x; d < D_MODEL; d += 256) { float v = xr[d]; ss += v * v; }
  red[threadIdx.x] = ss; __syncthreads();
  for (int o = 128; o > 0; o >>= 1) { if (threadIdx.x < o) red[threadIdx.x] += red[threadIdx.x + o]; __syncthreads(); }
  const float r = rsqrtf(red[0] / (float)D_MODEL + EPSF);
  for (int d = threadIdx.x; d < D_MODEL; d += 256)
    h[(size_t)s * D_MODEL + d] = (__bf16)(xr[d] * r * w[d]);
}

// ---------------- K2: fp32 -> bf16 cast (weights) ----------------
__global__ void cast_bf16_kernel(const float* __restrict__ w, __bf16* __restrict__ o, int n) {
  for (int i = blockIdx.x * blockDim.x + threadIdx.x; i < n; i += gridDim.x * blockDim.x)
    o[i] = (__bf16)w[i];
}

// ---------------- K3/K12: bf16 GEMM  out[M,N] = A[M,K] * W[N,K]^T (+resid) ----------------
// block = 256 threads = 8 waves; wave tile 16x64; block tile 64(M) x 128(N).
// A block-tile (64 x 64 per K-step) staged via TDM into double-buffered LDS.
__global__ void gemm_bf16_kernel(const __bf16* __restrict__ A, const __bf16* __restrict__ W,
                                 const float* __restrict__ resid, float* __restrict__ out,
                                 int M, int N, int K) {
  __shared__ __bf16 Abuf[2][64 * 64];     // 2 x 8 KB
  const int lane = threadIdx.x & 31;
  const int wave = threadIdx.x >> 5;
  const int lh = lane >> 4, ll = lane & 15;
  const int mblk = blockIdx.y * 64;
  const int m0 = (wave >> 1) * 16;        // row tile within 64-row block tile
  const int n0 = blockIdx.x * 128 + (wave & 1) * 64;
  const unsigned lds0 = (unsigned)(size_t)&Abuf[0][0];
  const unsigned lds1 = (unsigned)(size_t)&Abuf[1][0];
  const int nk = K >> 6;

  if (wave == 0)
    tdm_load_2d_bf16(lds0, A + (size_t)mblk * K, 64, 64, (unsigned)K, 64, (unsigned)K);

  v8f acc[4];
#pragma unroll
  for (int t = 0; t < 4; ++t) acc[t] = 0.0f;

  for (int kt = 0; kt < nk; ++kt) {
    if (wave == 0) {
      if (kt + 1 < nk) {
        tdm_load_2d_bf16((kt & 1) ? lds0 : lds1,
                         A + (size_t)mblk * K + (size_t)(kt + 1) * 64,
                         64, 64, (unsigned)K, 64, (unsigned)K);
        wait_tensorcnt1();                // tile kt done (TDM ops complete in order)
      } else {
        wait_tensorcnt0();
      }
    }
    __syncthreads();                      // publish tile kt to all waves
    const __bf16* abuf = Abuf[kt & 1];
    const int kg = kt * 64;
#pragma unroll
    for (int k2 = 0; k2 < 64; k2 += 32) {
      v16bf af = load_afrag(abuf + (m0 + ll) * 64 + k2, lh);
#pragma unroll
      for (int t = 0; t < 4; ++t) {
        const __bf16* wrow = W + (size_t)(n0 + 16 * t + ll) * K + kg + k2;
        acc[t] = wmma_bf16(af, load_bfrag(wrow, lh), acc[t]);
      }
    }
    __syncthreads();                      // all waves done reading before buffer reuse
  }
#pragma unroll
  for (int t = 0; t < 4; ++t)
#pragma unroll
    for (int r = 0; r < 8; ++r) {
      const int row = mblk + m0 + r + 8 * lh, col = n0 + 16 * t + ll;
      float v = acc[t][r];
      if (resid) v += resid[(size_t)row * N + col];
      out[(size_t)row * N + col] = v;
    }
}

// ---------------- K4: dt = softplus(proj_dt + bias), dA = -exp(A_log)*dt ----------------
__global__ void dt_kernel(const float* __restrict__ proj, const float* __restrict__ dt_bias,
                          const float* __restrict__ A_log, float* __restrict__ dt,
                          float* __restrict__ dA) {
  const int idx = blockIdx.x * blockDim.x + threadIdx.x;   // S*H
  const int s = idx >> 7, hh = idx & 127;
  float v = proj[(size_t)s * PROJ_DIM + (D_INNER + CONV_DIM) + hh] + dt_bias[hh];
  float sp = (v > 20.f) ? v : log1pf(__expf(v));
  dt[idx] = sp;
  dA[idx] = -__expf(A_log[hh]) * sp;
}

// ---------------- K5: causal conv(k=4) + SiLU, split into hid/B/C ----------------
__global__ void conv_kernel(const float* __restrict__ proj, const float* __restrict__ cw,
                            const float* __restrict__ cb, const float* __restrict__ dt,
                            float* __restrict__ hid, __bf16* __restrict__ hcb,
                            __bf16* __restrict__ Bb, __bf16* __restrict__ Cb) {
  const int idx = blockIdx.x * blockDim.x + threadIdx.x;   // S*CONV_DIM
  const int s = idx / CONV_DIM, ch = idx % CONV_DIM;
  float acc = cb[ch];
  const float* col = proj + D_INNER + ch;
#pragma unroll
  for (int i = 0; i < 4; ++i) {
    const int t = s - 3 + i;
    if (t >= 0) acc += col[(size_t)t * PROJ_DIM] * cw[ch * 4 + i];
  }
  const float v = acc * sigmoidf_(acc);
  if (ch < D_INNER) {
    hid[(size_t)s * D_INNER + ch] = v;
    hcb[(size_t)s * D_INNER + ch] = (__bf16)(v * dt[s * H_M + (ch >> 6)]);
  } else if (ch < D_INNER + GN) {
    Bb[(size_t)s * GN + (ch - D_INNER)] = (__bf16)v;
  } else {
    Cb[(size_t)s * GN + (ch - D_INNER - GN)] = (__bf16)v;
  }
}

// ---------------- K6: per-(chunk,head) cumsum of dA ----------------
__global__ void acs_kernel(const float* __restrict__ dA, float* __restrict__ Acs) {
  const int idx = blockIdx.x * blockDim.x + threadIdx.x;   // NCHUNK*H
  const int c = idx >> 7, hh = idx & 127;
  float a = 0.f;
  for (int l = 0; l < CHUNK; ++l) {
    a += dA[(size_t)(c * CHUNK + l) * H_M + hh];
    Acs[(size_t)idx * CHUNK + l] = a;
  }
}

// ---------------- K7: CB[c][g][i][j] = sum_n C[i,n]*B[j,n]  (WMMA) ----------------
__global__ void cb_kernel(const __bf16* __restrict__ Cb, const __bf16* __restrict__ Bb,
                          float* __restrict__ CB) {
  const int lane = threadIdx.x & 31, wave = threadIdx.x >> 5;
  const int lh = lane >> 4, ll = lane & 15;
  const int cg = blockIdx.y, c = cg >> 3, g = cg & 7;
  const int m0 = blockIdx.x * 64 + (wave >> 1) * 16;
  const int n0 = (wave & 1) * 64;
  const __bf16* Abase = Cb + (size_t)(c * CHUNK) * GN + g * N_M;
  const __bf16* Bbase = Bb + (size_t)(c * CHUNK) * GN + g * N_M;
  v8f acc[4];
#pragma unroll
  for (int t = 0; t < 4; ++t) acc[t] = 0.0f;
  for (int k = 0; k < N_M; k += 32) {
    v16bf af = load_afrag(Abase + (size_t)(m0 + ll) * GN + k, lh);
#pragma unroll
    for (int t = 0; t < 4; ++t)
      acc[t] = wmma_bf16(af, load_bfrag(Bbase + (size_t)(n0 + 16 * t + ll) * GN + k, lh), acc[t]);
  }
#pragma unroll
  for (int t = 0; t < 4; ++t)
#pragma unroll
    for (int r = 0; r < 8; ++r)
      CB[((size_t)cg * CHUNK + (m0 + r + 8 * lh)) * CHUNK + n0 + 16 * t + ll] = acc[t][r];
}

// ---------------- K8: states[c][h][p][n] = sum_l (hc*decay)[l,p] * B[l,n]  (WMMA via LDS) ----------------
__global__ void states_kernel(const __bf16* __restrict__ hcb, const __bf16* __restrict__ Bb,
                              const float* __restrict__ Acs, float* __restrict__ states) {
  __shared__ __bf16 hcdT[P_M * CHUNK];    // [p][l]
  __shared__ __bf16 BT[N_M * CHUNK];      // [n][l]
  __shared__ float  decay_s[CHUNK];
  const int ch = blockIdx.x;              // c*128 + head
  const int c = ch >> 7, hd = ch & 127, g = hd >> 4;
  const float* acs = Acs + (size_t)ch * CHUNK;
  if (threadIdx.x < 128) decay_s[threadIdx.x] = __expf(acs[127] - acs[threadIdx.x]);
  __syncthreads();
  for (int i = threadIdx.x; i < CHUNK * P_M; i += 256) {
    const int l = i >> 6, p = i & 63;
    float v = (float)hcb[(size_t)(c * CHUNK + l) * D_INNER + hd * P_M + p] * decay_s[l];
    hcdT[p * CHUNK + l] = (__bf16)v;
  }
  for (int i = threadIdx.x; i < CHUNK * N_M; i += 256) {
    const int l = i >> 7, n = i & 127;
    BT[n * CHUNK + l] = Bb[(size_t)(c * CHUNK + l) * GN + g * N_M + n];
  }
  __syncthreads();
  const int lane = threadIdx.x & 31, wave = threadIdx.x >> 5;
  const int lh = lane >> 4, ll = lane & 15;
  const int m0 = (wave >> 1) * 16;        // p tile
  const int n0 = (wave & 1) * 64;         // n tile
  v8f acc[4];
#pragma unroll
  for (int t = 0; t < 4; ++t) acc[t] = 0.0f;
  for (int k = 0; k < CHUNK; k += 32) {
    v16bf af = load_afrag(&hcdT[(m0 + ll) * CHUNK + k], lh);
#pragma unroll
    for (int t = 0; t < 4; ++t)
      acc[t] = wmma_bf16(af, load_bfrag(&BT[(n0 + 16 * t + ll) * CHUNK + k], lh), acc[t]);
  }
#pragma unroll
  for (int t = 0; t < 4; ++t)
#pragma unroll
    for (int r = 0; r < 8; ++r)
      states[((size_t)ch * P_M + (m0 + r + 8 * lh)) * N_M + n0 + 16 * t + ll] = acc[t][r];
}

// ---------------- K9: inter-chunk recurrence P_{i+1} = exp(Alast_i)*P_i + states_i ----------------
__global__ void recur_kernel(const float* __restrict__ states, const float* __restrict__ Acs,
                             __bf16* __restrict__ prevb) {
  const int idx = blockIdx.x * blockDim.x + threadIdx.x;   // H*P*N = 1M
  const int n = idx & 127, p = (idx >> 7) & 63, hd = idx >> 13;
  float pv = 0.f;
  for (int c = 0; c < NCHUNK; ++c) {
    const size_t ch = (size_t)c * H_M + hd;
    prevb[(ch * P_M + p) * N_M + n] = (__bf16)pv;
    const float alast = Acs[ch * CHUNK + 127];
    pv = __expf(alast) * pv + states[(ch * P_M + p) * N_M + n];
  }
}

// ---------------- K10: Y = (CB∘L)·hc + diag(exp(Acs))·(C·prev^T) + Dp*hid ----------------
__global__ void y_kernel(const float* __restrict__ CB, const __bf16* __restrict__ hcb,
                         const __bf16* __restrict__ Cb, const __bf16* __restrict__ prevb,
                         const float* __restrict__ Acs, const float* __restrict__ hid,
                         const float* __restrict__ Dp, float* __restrict__ y) {
  __shared__ __bf16 CBL[CHUNK * CHUNK];   // [i][j] masked+decayed
  __shared__ __bf16 hcT[P_M * CHUNK];     // [p][j]
  __shared__ float  acs_s[CHUNK];
  const int ch = blockIdx.x;              // c*128 + head
  const int c = ch >> 7, hd = ch & 127, g = hd >> 4;
  if (threadIdx.x < 128) acs_s[threadIdx.x] = Acs[(size_t)ch * CHUNK + threadIdx.x];
  __syncthreads();
  const float* cbp = CB + (size_t)(c * G_M + g) * CHUNK * CHUNK;
  for (int i = threadIdx.x; i < CHUNK * CHUNK; i += 256) {
    const int ii = i >> 7, jj = i & 127;
    float v = (ii >= jj) ? cbp[i] * __expf(acs_s[ii] - acs_s[jj]) : 0.f;
    CBL[i] = (__bf16)v;
  }
  for (int i = threadIdx.x; i < CHUNK * P_M; i += 256) {
    const int l = i >> 6, p = i & 63;
    hcT[p * CHUNK + l] = hcb[(size_t)(c * CHUNK + l) * D_INNER + hd * P_M + p];
  }
  __syncthreads();
  const int lane = threadIdx.x & 31, wave = threadIdx.x >> 5;
  const int lh = lane >> 4, ll = lane & 15;
  const int m0 = wave * 16;               // l (output row) tile
  v8f acc1[4], acc2[4];
#pragma unroll
  for (int t = 0; t < 4; ++t) { acc1[t] = 0.0f; acc2[t] = 0.0f; }
  for (int k = 0; k < CHUNK; k += 32) {
    v16bf a1 = load_afrag(&CBL[(m0 + ll) * CHUNK + k], lh);
    v16bf a2 = load_afrag(Cb + (size_t)(c * CHUNK + m0 + ll) * GN + g * N_M + k, lh);
#pragma unroll
    for (int t = 0; t < 4; ++t) {
      acc1[t] = wmma_bf16(a1, load_bfrag(&hcT[(16 * t + ll) * CHUNK + k], lh), acc1[t]);
      acc2[t] = wmma_bf16(a2, load_bfrag(prevb + ((size_t)ch * P_M + 16 * t + ll) * N_M + k, lh), acc2[t]);
    }
  }
  const float dpv = Dp[hd];
#pragma unroll
  for (int t = 0; t < 4; ++t)
#pragma unroll
    for (int r = 0; r < 8; ++r) {
      const int i = m0 + r + 8 * lh, p = 16 * t + ll;
      const int s = c * CHUNK + i;
      const size_t o = (size_t)s * D_INNER + hd * P_M + p;
      y[o] = acc1[t][r] + __expf(acs_s[i]) * acc2[t][r] + dpv * hid[o];
    }
}

// ---------------- K11: gate(SiLU) * y, per-group RMSNorm, *gnorm_w -> bf16 ----------------
__global__ void gate_norm_kernel(const float* __restrict__ y, const float* __restrict__ proj,
                                 const float* __restrict__ gw, __bf16* __restrict__ yfin) {
  __shared__ float red[256];
  const int s = blockIdx.x >> 3, g = blockIdx.x & 7;
  const int gsz = D_INNER / G_M;          // 1024
  float vals[4];
  float ss = 0.f;
#pragma unroll
  for (int j = 0; j < 4; ++j) {
    const int cidx = g * gsz + j * 256 + threadIdx.x;
    const float gv = proj[(size_t)s * PROJ_DIM + cidx];     // gate = first D_INNER cols
    const float yv = y[(size_t)s * D_INNER + cidx];
    const float t = yv * (gv * sigmoidf_(gv));
    vals[j] = t; ss += t * t;
  }
  red[threadIdx.x] = ss; __syncthreads();
  for (int o = 128; o > 0; o >>= 1) { if (threadIdx.x < o) red[threadIdx.x] += red[threadIdx.x + o]; __syncthreads(); }
  const float r = rsqrtf(red[0] / (float)gsz + EPSF);
#pragma unroll
  for (int j = 0; j < 4; ++j) {
    const int cidx = g * gsz + j * 256 + threadIdx.x;
    yfin[(size_t)s * D_INNER + cidx] = (__bf16)(vals[j] * r * gw[cidx]);
  }
}

// ---------------- launcher ----------------
extern "C" void kernel_launch(void* const* d_in, const int* in_sizes, int n_in,
                              void* d_out, int out_size, void* d_ws, size_t ws_size,
                              hipStream_t stream) {
  const float* x        = (const float*)d_in[0];
  const float* norm_w   = (const float*)d_in[1];
  const float* in_proj  = (const float*)d_in[2];
  const float* conv_w   = (const float*)d_in[3];
  const float* conv_b   = (const float*)d_in[4];
  const float* dt_bias  = (const float*)d_in[5];
  const float* A_log    = (const float*)d_in[6];
  const float* Dp       = (const float*)d_in[7];
  const float* gnorm_w  = (const float*)d_in[8];
  const float* out_proj = (const float*)d_in[9];
  float* out = (float*)d_out;

  char* ws = (char*)d_ws;
  size_t off = 0;
  auto alloc = [&](size_t bytes) { char* p = ws + off; off += (bytes + 255) & ~(size_t)255; return p; };

  __bf16* h_bf   = (__bf16*)alloc((size_t)S_LEN * D_MODEL * 2);          // 16.8 MB
  __bf16* wo_bf  = (__bf16*)alloc((size_t)D_MODEL * D_INNER * 2);        // 67.1 MB
  float*  proj   = (float*) alloc((size_t)S_LEN * PROJ_DIM * 4);         // 152 MB
  float*  dtb    = (float*) alloc((size_t)S_LEN * H_M * 4);
  float*  dAb    = (float*) alloc((size_t)S_LEN * H_M * 4);
  float*  hid    = (float*) alloc((size_t)S_LEN * D_INNER * 4);          // 67.1 MB
  __bf16* hcb    = (__bf16*)alloc((size_t)S_LEN * D_INNER * 2);          // 33.6 MB
  __bf16* Bb     = (__bf16*)alloc((size_t)S_LEN * GN * 2);
  __bf16* Cb     = (__bf16*)alloc((size_t)S_LEN * GN * 2);
  float*  ybuf   = (float*) alloc((size_t)S_LEN * D_INNER * 4);          // 67.1 MB
  __bf16* yfin   = (__bf16*)alloc((size_t)S_LEN * D_INNER * 2);          // 33.6 MB
  // wi_bf region (152 MB) — dead after GEMM1, overlaid by SSD intermediates
  char*   region = alloc((size_t)PROJ_DIM * D_MODEL * 2);
  __bf16* wi_bf  = (__bf16*)region;
  float*  Acs    = (float*) (region);                                    // 1.05 MB
  float*  CBm    = (float*) (region + ((size_t)1 << 21));                // 8.4 MB
  float*  states = (float*) (region + ((size_t)11 << 20));               // 67.1 MB
  __bf16* prevb  = (__bf16*)(region + ((size_t)80 << 20));               // 33.6 MB
  (void)in_sizes; (void)n_in; (void)out_size; (void)ws_size;

  // 1. RMSNorm + bf16 cast
  rmsnorm_cast_kernel<<<S_LEN, 256, 0, stream>>>(x, norm_w, h_bf);
  // 2. weight casts
  cast_bf16_kernel<<<8192, 256, 0, stream>>>(in_proj, wi_bf, PROJ_DIM * D_MODEL);
  cast_bf16_kernel<<<8192, 256, 0, stream>>>(out_proj, wo_bf, D_MODEL * D_INNER);
  // 3. in_proj GEMM: proj[2048,18560] = h * Wi^T   (TDM-staged A tiles)
  gemm_bf16_kernel<<<dim3(PROJ_DIM / 128, S_LEN / 64), 256, 0, stream>>>(
      h_bf, wi_bf, nullptr, proj, S_LEN, PROJ_DIM, D_MODEL);
  // 4. dt / dA
  dt_kernel<<<(S_LEN * H_M) / 256, 256, 0, stream>>>(proj, dt_bias, A_log, dtb, dAb);
  // 5. conv + SiLU + split
  conv_kernel<<<(S_LEN * CONV_DIM) / 256, 256, 0, stream>>>(proj, conv_w, conv_b, dtb,
                                                            hid, hcb, Bb, Cb);
  // 6. cumsum(dA) per (chunk, head)
  acs_kernel<<<(NCHUNK * H_M) / 256, 256, 0, stream>>>(dAb, Acs);
  // 7. CB = C * B^T per (chunk, group)
  cb_kernel<<<dim3(2, NCHUNK * G_M), 256, 0, stream>>>(Cb, Bb, CBm);
  // 8. chunk states
  states_kernel<<<NCHUNK * H_M, 256, 0, stream>>>(hcb, Bb, Acs, states);
  // 9. inter-chunk recurrence -> prev states (bf16)
  recur_kernel<<<(H_M * P_M * N_M) / 256, 256, 0, stream>>>(states, Acs, prevb);
  // 10. Y = diag + off + D residual
  y_kernel<<<NCHUNK * H_M, 256, 0, stream>>>(CBm, hcb, Cb, prevb, Acs, hid, Dp, ybuf);
  // 11. gate + group RMSNorm -> bf16
  gate_norm_kernel<<<S_LEN * G_M, 256, 0, stream>>>(ybuf, proj, gnorm_w, yfin);
  // 12. out_proj GEMM + residual
  gemm_bf16_kernel<<<dim3(D_MODEL / 128, S_LEN / 64), 256, 0, stream>>>(
      yfin, wo_bf, x, out, S_LEN, D_MODEL, D_INNER);
}
